// Mamba2Block_68161130987996
// MI455X (gfx1250) — compile-verified, matrix operand order checked by reading, not run on hardware
//
#include <hip/hip_runtime.h>
#include <math.h>

typedef __attribute__((ext_vector_type(16))) __bf16 v16bf;
typedef __attribute__((ext_vector_type(2)))  __bf16 v2bf;
typedef __attribute__((ext_vector_type(8)))  float  v8f;

#define BATCH   2
#define SEQ     2048
#define NTOK    (BATCH * SEQ)          // 4096
#define DMODEL  1024
#define DINNER  2048
#define HSSM    32
#define PDIM    64
#define NSTATE  128
#define DCONV   4
#define CHUNKL  256
#define NCHUNK  (SEQ / CHUNKL)         // 8
#define CONVDIM (DINNER + 2 * NSTATE)  // 2304
#define DINPROJ (2 * DINNER + 2 * NSTATE + HSSM) // 4384
#define EPSVAL  1e-5f

// ---------- bf16 helpers (round-to-nearest-even) ----------
__device__ __forceinline__ unsigned short f2bfu(float f) {
  union { float f; unsigned u; } in; in.f = f;
  unsigned r = in.u + 0x7FFFu + ((in.u >> 16) & 1u);
  return (unsigned short)(r >> 16);
}
__device__ __forceinline__ __bf16 u2bf(unsigned short s) {
  union { unsigned short s; __bf16 b; } o; o.s = s; return o.b;
}
__device__ __forceinline__ float u2f(unsigned short s) {
  union { unsigned u; float f; } o; o.u = ((unsigned)s) << 16; return o.f;
}
// pack two fp32 -> packed bf16x2 (single v_cvt_pk_bf16_f32 when available)
__device__ __forceinline__ unsigned packbf2(float lo, float hi) {
#if __has_builtin(__builtin_amdgcn_cvt_pk_bf16_f32)
  union { v2bf v; unsigned u; } o;
  o.v = __builtin_amdgcn_cvt_pk_bf16_f32(lo, hi);
  return o.u;
#else
  return ((unsigned)f2bfu(hi) << 16) | (unsigned)f2bfu(lo);
#endif
}
// K index inside a 16x32 bf16 WMMA fragment (ISA 16-bit A-matrix layout):
// element e of lane L covers K = 16*(e/8) + 8*(L>=16) + (e%8)
__device__ __forceinline__ int wkidx(int e, int hi8) {
  return ((e >> 3) << 4) + hi8 + (e & 7);
}
// 16x32 fragment from packed-bf16 memory; lane row = lane&15, contiguous K.
__device__ __forceinline__ v16bf frag_rows_bf16(const unsigned short* base,
                                                int rstride, int lane) {
  const unsigned short* p = base + (size_t)(lane & 15) * rstride;
  const int hi8 = (lane >> 4) << 3;
  union { v16bf v; unsigned u[8]; } o;
#pragma unroll
  for (int i = 0; i < 8; ++i)
    o.u[i] = *(const unsigned*)(p + wkidx(i * 2, hi8));  // pairs are adjacent
  return o.v;
}
// 16x32 fragment from fp32 memory (paired packed conversion).
__device__ __forceinline__ v16bf frag_rows_f32(const float* base, int rstride,
                                               int lane) {
  const float* p = base + (size_t)(lane & 15) * rstride;
  const int hi8 = (lane >> 4) << 3;
  union { v16bf v; unsigned u[8]; } o;
#pragma unroll
  for (int i = 0; i < 8; ++i) {
    int k = wkidx(i * 2, hi8);
    o.u[i] = packbf2(p[k], p[k + 1]);
  }
  return o.v;
}

// ---------------- fp32 -> bf16 conversion pass (n multiple of 4) -----------
__global__ __launch_bounds__(256) void k_cvt_bf16(
    const float* __restrict__ src, unsigned short* __restrict__ dst, int n4) {
  int i = blockIdx.x * 256 + threadIdx.x;
  if (i >= n4) return;
  float4 v = ((const float4*)src)[i];
  uint2 o;
  o.x = packbf2(v.x, v.y);
  o.y = packbf2(v.z, v.w);
  ((uint2*)dst)[i] = o;
}

// ---------------- bf16 WMMA GEMM, async double-buffered tiles --------------
// A: bf16 [M x K] row-major, W: bf16 [N x K] row-major, out fp32 [M x N].
#define GBM 128
#define GBN 128
#define GBK 32
#define GLD 40                    // LDS row stride (shorts): 64B data+16B pad
#define TILEB (GBM * GLD * 2)     // one tile buffer in bytes

__global__ __launch_bounds__(256) void k_gemm_bf16(
    const unsigned short* __restrict__ A, const unsigned short* __restrict__ W,
    const float* __restrict__ resid, float* __restrict__ out,
    int M, int N, int K) {
  __shared__ unsigned short sA[2][GBM][GLD];
  __shared__ unsigned short sB[2][GBN][GLD];
  const int tid  = threadIdx.x;
  const int lane = tid & 31;
  const int wave = tid >> 5;
  const int col  = lane & 15;
  const int hi8  = (lane >> 4) << 3;
  const int wm   = (wave & 3) * 32;   // 4 waves along M
  const int wn   = (wave >> 2) * 64;  // 2 waves along N
  const int m0   = blockIdx.y * GBM;
  const int n0   = blockIdx.x * GBN;
  const unsigned ldsA = (unsigned)(size_t)(&sA[0][0][0]);
  const unsigned ldsB = (unsigned)(size_t)(&sB[0][0][0]);

  // issue one tile's async copies: 4 async instructions per wave
  auto fill_tile = [&](int pb, int k0) {
#pragma unroll
    for (int i = 0; i < 2; ++i) {     // 512 16B segments per tile pair
      int seg = tid + i * 256;
      int r = seg >> 2, part = seg & 3;
      int n = n0 + r; if (n >= N) n = N - 1;  // clamp; cols >= N never stored
      const unsigned short* gA = A + (size_t)(m0 + r) * K + k0 + part * 8;
      const unsigned short* gB = W + (size_t)n * K + k0 + part * 8;
      unsigned la = ldsA + (unsigned)(pb * TILEB + r * (GLD * 2) + part * 16);
      unsigned lb = ldsB + (unsigned)(pb * TILEB + r * (GLD * 2) + part * 16);
      asm volatile("global_load_async_to_lds_b128 %0, %1, off"
                   :: "v"(la), "v"((unsigned long long)(size_t)gA) : "memory");
      asm volatile("global_load_async_to_lds_b128 %0, %1, off"
                   :: "v"(lb), "v"((unsigned long long)(size_t)gB) : "memory");
    }
  };

  v8f acc[2][4];
#pragma unroll
  for (int i = 0; i < 2; ++i)
#pragma unroll
    for (int j = 0; j < 4; ++j) acc[i][j] = (v8f)(0.0f);

  const int nk = K / GBK;
  int pb = 0;
  fill_tile(0, 0);
  for (int kt = 0; kt < nk; ++kt) {
    if (kt + 1 < nk) {
      fill_tile(pb ^ 1, (kt + 1) * GBK);      // prefetch next tile
      asm volatile("s_wait_asynccnt 4" ::: "memory");  // current tile landed
    } else {
      asm volatile("s_wait_asynccnt 0" ::: "memory");
    }
    __syncthreads();
    v16bf af[2], bfr[4];
#pragma unroll
    for (int mi = 0; mi < 2; ++mi) {
      int r = wm + mi * 16 + col;
#pragma unroll
      for (int e = 0; e < 16; ++e) af[mi][e] = u2bf(sA[pb][r][wkidx(e, hi8)]);
    }
#pragma unroll
    for (int ni = 0; ni < 4; ++ni) {
      int r = wn + ni * 16 + col;
#pragma unroll
      for (int e = 0; e < 16; ++e) bfr[ni][e] = u2bf(sB[pb][r][wkidx(e, hi8)]);
    }
#pragma unroll
    for (int mi = 0; mi < 2; ++mi)
#pragma unroll
      for (int ni = 0; ni < 4; ++ni)
        acc[mi][ni] = __builtin_amdgcn_wmma_f32_16x16x32_bf16(
            false, af[mi], false, bfr[ni], (short)0, acc[mi][ni], false, false);
    __syncthreads();   // all waves done with tile pb before it is refilled
    pb ^= 1;
  }
#pragma unroll
  for (int mi = 0; mi < 2; ++mi)
#pragma unroll
    for (int ni = 0; ni < 4; ++ni) {
      int cg = n0 + wn + ni * 16 + col;
      if (cg >= N) continue;
#pragma unroll
      for (int v = 0; v < 8; ++v) {
        int rg = m0 + wm + mi * 16 + v + hi8;  // D row = v + 8*(lane>=16)
        float val = acc[mi][ni][v];
        if (resid) val += resid[(size_t)rg * N + cg];
        out[(size_t)rg * N + cg] = val;
      }
    }
}

// ---------------- RoPE (in place, fp32) + RMSNorm -> h (bf16) --------------
__global__ __launch_bounds__(256) void k_rope_rmsnorm(
    float* __restrict__ resid, unsigned short* __restrict__ hout,
    const float* __restrict__ norm_w) {
  __shared__ float sq[DMODEL];
  __shared__ float red[256];
  const int tid = threadIdx.x;
  const size_t tok = blockIdx.x;
  const int tpos = (int)(tok % SEQ);
  for (int i = tid; i < DMODEL; i += 256) sq[i] = resid[tok * DMODEL + i];
  __syncthreads();
  float vals[4]; float ss = 0.f;
#pragma unroll
  for (int i = 0; i < 4; ++i) {
    int d = tid + i * 256;
    int r = d & 63;
    int j = (r < 32) ? r : r - 32;
    float ang = (float)tpos * __powf(10000.0f, -(float)(2 * j) * (1.0f / 64.0f));
    float cs = __cosf(ang), sn = __sinf(ang);
    float v = (r < 32) ? (sq[d] * cs - sq[d + 32] * sn)
                       : (sq[d] * cs + sq[d - 32] * sn);
    vals[i] = v; ss += v * v;
  }
  red[tid] = ss; __syncthreads();
  for (int o = 128; o > 0; o >>= 1) {
    if (tid < o) red[tid] += red[tid + o];
    __syncthreads();
  }
  float scale = rsqrtf(red[0] * (1.0f / (float)DMODEL) + EPSVAL);
#pragma unroll
  for (int i = 0; i < 4; ++i) {
    int d = tid + i * 256;
    resid[tok * DMODEL + d] = vals[i];
    hout[tok * DMODEL + d] = f2bfu(vals[i] * scale * norm_w[d]);
  }
}

// ---------------- causal conv1d(4) + SiLU -> bf16 --------------------------
__global__ __launch_bounds__(256) void k_conv_silu(
    const float* __restrict__ zx, const float* __restrict__ cw,
    const float* __restrict__ cb, unsigned short* __restrict__ convo) {
  size_t idx = (size_t)blockIdx.x * 256 + threadIdx.x;
  if (idx >= (size_t)NTOK * CONVDIM) return;
  int ch = (int)(idx % CONVDIM);
  size_t tk = idx / CONVDIM;
  int t = (int)(tk % SEQ);
  size_t bb = tk / SEQ;
  float acc = cb[ch];
#pragma unroll
  for (int k = 0; k < DCONV; ++k) {
    int tt = t + k - (DCONV - 1);
    if (tt >= 0)
      acc += zx[((size_t)bb * SEQ + tt) * DINPROJ + DINNER + ch] * cw[ch * DCONV + k];
  }
  convo[idx] = f2bfu(acc / (1.0f + __expf(-acc)));  // SiLU, bf16
}

// ---------------- dt = softplus(raw + bias) --------------------------------
__global__ __launch_bounds__(256) void k_dt(
    const float* __restrict__ zx, const float* __restrict__ dt_bias,
    float* __restrict__ dtb) {
  int idx = blockIdx.x * 256 + threadIdx.x;
  if (idx >= NTOK * HSSM) return;
  int h = idx % HSSM;
  int tk = idx / HSSM;
  float v = zx[(size_t)tk * DINPROJ + (DINPROJ - HSSM) + h] + dt_bias[h];
  dtb[idx] = (v > 20.0f) ? v : log1pf(__expf(v));
}

// ---------------- SSD intra-chunk: Y_diag + D*x, chunk states --------------
__global__ __launch_bounds__(256) void k_ssd_chunk(
    const unsigned short* __restrict__ convo, const float* __restrict__ dtb,
    const float* __restrict__ A_log, const float* __restrict__ D_param,
    float* __restrict__ Y, float* __restrict__ states,
    float* __restrict__ acsO, float* __restrict__ atotO) {
  __shared__ float sAdt[CHUNKL];
  __shared__ float sAcs[CHUNKL];
  __shared__ float sDec[CHUNKL];
  __shared__ float sDt[CHUNKL];
  __shared__ unsigned short sX[CHUNKL][PDIM];   // xdt bf16 (later xdt*decay)
  __shared__ unsigned short sCB[8][16][36];     // per-wave masked CB tile

  const int h = blockIdx.x, c = blockIdx.y, b = blockIdx.z;
  const int tid = threadIdx.x, lane = tid & 31, wave = tid >> 5;
  const int col = lane & 15, hi8 = (lane >> 4) << 3;
  const size_t tok0 = (size_t)b * SEQ + (size_t)c * CHUNKL;

  {
    float dtv = dtb[(tok0 + tid) * HSSM + h];
    float Ah = -__expf(A_log[h]);
    sDt[tid] = dtv;
    sAdt[tid] = dtv * Ah;
  }
  __syncthreads();
  sAcs[tid] = sAdt[tid];
  __syncthreads();
  for (int off = 1; off < CHUNKL; off <<= 1) {   // inclusive scan
    float v = sAcs[tid];
    float add = (tid >= off) ? sAcs[tid - off] : 0.0f;
    __syncthreads();
    sAcs[tid] = v + add;
    __syncthreads();
  }
  float total = sAcs[CHUNKL - 1];
  sDec[tid] = __expf(total - sAcs[tid]);
  acsO[((size_t)b * HSSM + h) * SEQ + (size_t)c * CHUNKL + tid] = sAcs[tid];
  if (tid == 0) atotO[((size_t)b * HSSM + h) * NCHUNK + c] = total;
  for (int i = tid; i < CHUNKL * PDIM / 2; i += 256) {  // xdt = x * dt (bf16)
    int l = i >> 5, p2 = (i & 31) * 2;
    unsigned pr = *(const unsigned*)(convo + (tok0 + l) * CONVDIM + h * PDIM + p2);
    float d = sDt[l];
    *(unsigned*)(&sX[l][p2]) =
        packbf2(u2f((unsigned short)pr) * d, u2f((unsigned short)(pr >> 16)) * d);
  }
  __syncthreads();

  const unsigned short* Cbase = convo + tok0 * CONVDIM + DINNER + NSTATE;
  const unsigned short* Bbase = convo + tok0 * CONVDIM + DINNER;
  const float Dh = D_param[h];

  // ---- Y_diag = (exp(segsum) ⊙ C·Bᵀ) · xdt,  two l-tiles per wave ----
  for (int lt = wave; lt < CHUNKL / 16; lt += 8) {
    const int l0 = lt * 16;
    v8f yacc[4];
#pragma unroll
    for (int p = 0; p < 4; ++p) yacc[p] = (v8f)(0.0f);
    const int n32 = ((l0 + 15) >> 5) + 1;  // causal s-range in 32-steps
    for (int s32 = 0; s32 < n32; ++s32) {
      const int s0 = s32 * 32;
#pragma unroll
      for (int sub = 0; sub < 2; ++sub) {
        v8f cb = (v8f)(0.0f);
#pragma unroll
        for (int kk = 0; kk < 4; ++kk) {  // K = n = 128
          v16bf aC = frag_rows_bf16(Cbase + (size_t)l0 * CONVDIM + kk * 32,
                                    CONVDIM, lane);
          v16bf bB = frag_rows_bf16(Bbase + (size_t)(s0 + 16 * sub) * CONVDIM + kk * 32,
                                    CONVDIM, lane);
          cb = __builtin_amdgcn_wmma_f32_16x16x32_bf16(
              false, aC, false, bB, (short)0, cb, false, false);
        }
#pragma unroll
        for (int v = 0; v < 8; ++v) {     // causal mask + decay, reshape via LDS
          int li = l0 + v + hi8;
          int si = s0 + 16 * sub + col;
          float val = (si <= li) ? cb[v] * __expf(sAcs[li] - sAcs[si]) : 0.0f;
          sCB[wave][v + hi8][16 * sub + col] = f2bfu(val);
        }
      }
      v16bf a2;
#pragma unroll
      for (int e = 0; e < 16; ++e) a2[e] = u2bf(sCB[wave][col][wkidx(e, hi8)]);
#pragma unroll
      for (int pt = 0; pt < 4; ++pt) {
        v16bf bx;
#pragma unroll
        for (int e = 0; e < 16; ++e)
          bx[e] = u2bf(sX[s0 + wkidx(e, hi8)][pt * 16 + col]);
        yacc[pt] = __builtin_amdgcn_wmma_f32_16x16x32_bf16(
            false, a2, false, bx, (short)0, yacc[pt], false, false);
      }
    }
#pragma unroll
    for (int pt = 0; pt < 4; ++pt)
#pragma unroll
      for (int v = 0; v < 8; ++v) {
        int l = l0 + v + hi8;
        int p = pt * 16 + col;
        float xv = u2f(convo[(tok0 + l) * CONVDIM + h * PDIM + p]);
        Y[((tok0 + l) * HSSM + h) * PDIM + p] = yacc[pt][v] + Dh * xv;
      }
  }

  // ---- in-place: sX <- xdt * decay (after all Y_diag reads) ----
  __syncthreads();
  for (int i = tid; i < CHUNKL * PDIM / 2; i += 256) {
    int l = i >> 5, p2 = (i & 31) * 2;
    unsigned pr = *(unsigned*)(&sX[l][p2]);
    float d = sDec[l];
    *(unsigned*)(&sX[l][p2]) =
        packbf2(u2f((unsigned short)pr) * d, u2f((unsigned short)(pr >> 16)) * d);
  }
  __syncthreads();

  // ---- chunk states  S[p][n] = Σ_l (decay·xdt)[l][p]·B[l][n]  ----
#pragma unroll
  for (int i = 0; i < 4; ++i) {           // 32 tiles (4x8) over 8 waves
    int tile = wave * 4 + i;
    int mt = tile >> 3, nt = tile & 7;
    v8f st = (v8f)(0.0f);
    for (int kk = 0; kk < 8; ++kk) {      // K = l = 256
      int k0 = kk * 32;
      v16bf aX, bN;
#pragma unroll
      for (int e = 0; e < 16; ++e)
        aX[e] = u2bf(sX[k0 + wkidx(e, hi8)][mt * 16 + col]);
      const unsigned short* bp = Bbase + (size_t)k0 * CONVDIM + nt * 16 + col;
#pragma unroll
      for (int e = 0; e < 16; ++e)
        bN[e] = u2bf(bp[(size_t)wkidx(e, hi8) * CONVDIM]);
      st = __builtin_amdgcn_wmma_f32_16x16x32_bf16(
          false, aX, false, bN, (short)0, st, false, false);
    }
    float* sp = states + (((size_t)b * NCHUNK + c) * HSSM + h) * (PDIM * NSTATE);
#pragma unroll
    for (int v = 0; v < 8; ++v)
      sp[(size_t)(mt * 16 + v + hi8) * NSTATE + nt * 16 + col] = st[v];
  }
}

// ---------------- inter-chunk state scan:  P_{c+1} = e^{A_c} P_c + S_c -----
__global__ __launch_bounds__(256) void k_state_scan(
    const float* __restrict__ states, const float* __restrict__ atot,
    float* __restrict__ prev) {
  const int bh = blockIdx.x;
  const int b = bh / HSSM, h = bh % HSSM;
  const int tid = threadIdx.x;
  float P[32];
#pragma unroll
  for (int i = 0; i < 32; ++i) P[i] = 0.0f;
  for (int c = 0; c < NCHUNK; ++c) {
    size_t base = (((size_t)b * NCHUNK + c) * HSSM + h) * (PDIM * NSTATE);
    float a = __expf(atot[((size_t)b * HSSM + h) * NCHUNK + c]);
#pragma unroll
    for (int i = 0; i < 32; ++i) {
      int idx = tid + i * 256;
      prev[base + idx] = P[i];                       // state entering chunk c
      P[i] = a * P[i] + states[base + idx];
    }
  }
}

// ---------------- SSD inter-chunk:  Y += diag(e^{Acs}) · C · Pᵀ ------------
__global__ __launch_bounds__(256) void k_ssd_yoff(
    const unsigned short* __restrict__ convo, const float* __restrict__ prev,
    const float* __restrict__ acs, float* __restrict__ Y) {
  __shared__ float sAcs[CHUNKL];
  const int h = blockIdx.x, c = blockIdx.y, b = blockIdx.z;
  const int tid = threadIdx.x, lane = tid & 31, wave = tid >> 5;
  const int col = lane & 15, hi8 = (lane >> 4) << 3;
  const size_t tok0 = (size_t)b * SEQ + (size_t)c * CHUNKL;
  sAcs[tid] = acs[((size_t)b * HSSM + h) * SEQ + (size_t)c * CHUNKL + tid];
  __syncthreads();
  const unsigned short* Cbase = convo + tok0 * CONVDIM + DINNER + NSTATE;
  const float* Pbase = prev + (((size_t)b * NCHUNK + c) * HSSM + h) * (PDIM * NSTATE);
  for (int lt = wave; lt < CHUNKL / 16; lt += 8) {
    const int l0 = lt * 16;
    v8f yacc[4];
#pragma unroll
    for (int p = 0; p < 4; ++p) yacc[p] = (v8f)(0.0f);
    for (int kk = 0; kk < 4; ++kk) {  // K = n = 128
      v16bf aC = frag_rows_bf16(Cbase + (size_t)l0 * CONVDIM + kk * 32,
                                CONVDIM, lane);
#pragma unroll
      for (int pt = 0; pt < 4; ++pt) {
        v16bf bp = frag_rows_f32(Pbase + (size_t)(pt * 16) * NSTATE + kk * 32,
                                 NSTATE, lane);
        yacc[pt] = __builtin_amdgcn_wmma_f32_16x16x32_bf16(
            false, aC, false, bp, (short)0, yacc[pt], false, false);
      }
    }
    float rowsc[8];                   // decay applied on fp32 output rows
#pragma unroll
    for (int v = 0; v < 8; ++v) rowsc[v] = __expf(sAcs[l0 + v + hi8]);
#pragma unroll
    for (int pt = 0; pt < 4; ++pt)
#pragma unroll
      for (int v = 0; v < 8; ++v) {
        int l = l0 + v + hi8;
        int p = pt * 16 + col;
        Y[((tok0 + l) * HSSM + h) * PDIM + p] += yacc[pt][v] * rowsc[v];
      }
  }
}

// ---------------- gated RMSNorm -> bf16 ------------------------------------
__global__ __launch_bounds__(256) void k_gated_norm(
    const float* __restrict__ Y, const float* __restrict__ zx,
    const float* __restrict__ gw, unsigned short* __restrict__ ygb) {
  __shared__ float red[256];
  const int tid = threadIdx.x;
  const size_t tok = blockIdx.x;
  float g[8]; float ss = 0.0f;
#pragma unroll
  for (int i = 0; i < 8; ++i) {
    int d = tid + i * 256;
    float y = Y[tok * DINNER + d];
    float z = zx[tok * DINPROJ + d];
    float gv = y * (z / (1.0f + __expf(-z)));
    g[i] = gv; ss += gv * gv;
  }
  red[tid] = ss; __syncthreads();
  for (int o = 128; o > 0; o >>= 1) {
    if (tid < o) red[tid] += red[tid + o];
    __syncthreads();
  }
  float scale = rsqrtf(red[0] * (1.0f / (float)DINNER) + EPSVAL);
#pragma unroll
  for (int i = 0; i < 8; ++i) {
    int d = tid + i * 256;
    ygb[tok * DINNER + d] = f2bfu(g[i] * scale * gw[d]);
  }
}

// ---------------------------------------------------------------------------
extern "C" void kernel_launch(void* const* d_in, const int* in_sizes, int n_in,
                              void* d_out, int out_size, void* d_ws, size_t ws_size,
                              hipStream_t stream) {
  const float* x        = (const float*)d_in[0];
  const float* W_in     = (const float*)d_in[1];
  const float* W_zxbcdt = (const float*)d_in[2];
  const float* conv_w   = (const float*)d_in[3];
  const float* conv_b   = (const float*)d_in[4];
  const float* dt_bias  = (const float*)d_in[5];
  const float* A_log    = (const float*)d_in[6];
  const float* D_param  = (const float*)d_in[7];
  const float* norm_pre = (const float*)d_in[8];
  const float* norm_gat = (const float*)d_in[9];
  const float* W_out    = (const float*)d_in[10];
  float* out = (float*)d_out;

  float* ws = (float*)d_ws;
  size_t off = 0;
  float* resid = ws + off; off += (size_t)NTOK * DMODEL;
  float* zx    = ws + off; off += (size_t)NTOK * DINPROJ;
  float* dtb   = ws + off; off += (size_t)NTOK * HSSM;
  float* Ybuf  = ws + off; off += (size_t)NTOK * DINNER;
  float* st    = ws + off; off += (size_t)BATCH * NCHUNK * HSSM * PDIM * NSTATE;
  float* pv    = ws + off; off += (size_t)BATCH * NCHUNK * HSSM * PDIM * NSTATE;
  float* acs   = ws + off; off += (size_t)BATCH * HSSM * SEQ;
  float* atot  = ws + off; off += (size_t)BATCH * HSSM * NCHUNK;
  // bf16 buffers (2 bytes/elem, counted in floats)
  unsigned short* convo = (unsigned short*)(ws + off); off += (size_t)NTOK * CONVDIM / 2;
  unsigned short* xb  = (unsigned short*)(ws + off); off += (size_t)NTOK * DMODEL / 2;
  unsigned short* hb  = (unsigned short*)(ws + off); off += (size_t)NTOK * DMODEL / 2;
  unsigned short* ygb = (unsigned short*)(ws + off); off += (size_t)NTOK * DINNER / 2;
  unsigned short* wib = (unsigned short*)(ws + off); off += (size_t)DMODEL * DMODEL / 2;
  unsigned short* wzb = (unsigned short*)(ws + off); off += (size_t)DINPROJ * DMODEL / 2;
  unsigned short* wob = (unsigned short*)(ws + off); off += (size_t)DMODEL * DINNER / 2;

  // 0. one-shot fp32->bf16 conversions (amortized over 8-32x tile reuse)
  {
    int n4;
    n4 = NTOK * DMODEL / 4;
    k_cvt_bf16<<<(n4 + 255) / 256, 256, 0, stream>>>(x, xb, n4);
    n4 = DMODEL * DMODEL / 4;  // only q-rows of W_in are used downstream
    k_cvt_bf16<<<(n4 + 255) / 256, 256, 0, stream>>>(W_in, wib, n4);
    n4 = DINPROJ * DMODEL / 4;
    k_cvt_bf16<<<(n4 + 255) / 256, 256, 0, stream>>>(W_zxbcdt, wzb, n4);
    n4 = DMODEL * DINNER / 4;
    k_cvt_bf16<<<(n4 + 255) / 256, 256, 0, stream>>>(W_out, wob, n4);
  }
  // 1. q = x @ W_in[0:1024,:]^T
  k_gemm_bf16<<<dim3(DMODEL / GBN, NTOK / GBM), 256, 0, stream>>>(
      xb, wib, nullptr, resid, NTOK, DMODEL, DMODEL);
  // 2. RoPE (in place -> residual) + pre-RMSNorm -> h (bf16)
  k_rope_rmsnorm<<<NTOK, 256, 0, stream>>>(resid, hb, norm_pre);
  // 3. zxbcdt = h @ W_zxbcdt^T  (dominant GEMM, N=4384)
  k_gemm_bf16<<<dim3((DINPROJ + GBN - 1) / GBN, NTOK / GBM), 256, 0, stream>>>(
      hb, wzb, nullptr, zx, NTOK, DINPROJ, DMODEL);
  // 4. causal conv + SiLU -> bf16 ; dt = softplus
  k_conv_silu<<<(int)(((size_t)NTOK * CONVDIM + 255) / 256), 256, 0, stream>>>(
      zx, conv_w, conv_b, convo);
  k_dt<<<(NTOK * HSSM + 255) / 256, 256, 0, stream>>>(zx, dt_bias, dtb);
  // 5. SSD: intra-chunk, state scan, inter-chunk
  k_ssd_chunk<<<dim3(HSSM, NCHUNK, BATCH), 256, 0, stream>>>(
      convo, dtb, A_log, D_param, Ybuf, st, acs, atot);
  k_state_scan<<<BATCH * HSSM, 256, 0, stream>>>(st, atot, pv);
  k_ssd_yoff<<<dim3(HSSM, NCHUNK, BATCH), 256, 0, stream>>>(convo, pv, acs, Ybuf);
  // 6. gated RMSNorm -> bf16, out = yg @ W_out^T + residual
  k_gated_norm<<<NTOK, 256, 0, stream>>>(Ybuf, zx, norm_gat, ygb);
  k_gemm_bf16<<<dim3(DMODEL / GBN, NTOK / GBM), 256, 0, stream>>>(
      ygb, wob, resid, out, NTOK, DMODEL, DINNER);
}
